// Custom_GNN_56650618634427
// MI455X (gfx1250) — matrix operand order repaired and checked by reading, not compiled, not run on hardware
//
#include <hip/hip_runtime.h>
#include <hip/hip_bf16.h>
#include <math.h>

// ---------------------------------------------------------------------------
// Types for CDNA5 WMMA (wave32): v_wmma_f32_16x16x32_bf16
// ---------------------------------------------------------------------------
typedef __attribute__((ext_vector_type(16))) __bf16 v16bf;
typedef __attribute__((ext_vector_type(8)))  float  v8f;
typedef __attribute__((ext_vector_type(4)))  int    v4i;

#define GAS __attribute__((address_space(1)))
#define LAS __attribute__((address_space(3)))

#if __has_builtin(__builtin_amdgcn_global_load_async_to_lds_b128)
#define USE_ASYNC_LDS 1
#else
#define USE_ASYNC_LDS 0
#endif

// global_load_async_to_lds_b128 signature (from hipcc diagnostic):
//   (int4 AS1* src, int4 AS3* dst, imm int offset, imm int cpol)
__device__ __forceinline__ GAS v4i* to_gas4(const void* p) {
  return (GAS v4i*)(uintptr_t)p;
}
__device__ __forceinline__ LAS v4i* to_las4(void* p) {
  return (LAS v4i*)(unsigned)(uintptr_t)p;   // LDS offset lives in addr[31:0]
}
__device__ __forceinline__ void wait_async0() {
#if USE_ASYNC_LDS
#if __has_builtin(__builtin_amdgcn_s_wait_asynccnt)
  __builtin_amdgcn_s_wait_asynccnt(0);
#else
  asm volatile("s_wait_asynccnt 0x0" ::: "memory");
#endif
#endif
}

__device__ __forceinline__ float bf2f(unsigned v) {
  unsigned u = v << 16;
  return __builtin_bit_cast(float, u);
}
__device__ __forceinline__ unsigned short f2bf(float f) {
  unsigned x = __builtin_bit_cast(unsigned, f);
  x += 0x7FFFu + ((x >> 16) & 1u);            // round-to-nearest-even
  return (unsigned short)(x >> 16);
}
__device__ __forceinline__ float wred_sum(float v) {
#pragma unroll
  for (int o = 16; o; o >>= 1) v += __shfl_xor(v, o, 32);
  return v;
}
__device__ __forceinline__ float wred_max(float v) {
#pragma unroll
  for (int o = 16; o; o >>= 1) v = fmaxf(v, __shfl_xor(v, o, 32));
  return v;
}
__device__ __forceinline__ float dot8_bf16(uint4 a, uint4 b) {
  const unsigned* pa = (const unsigned*)&a;
  const unsigned* pb = (const unsigned*)&b;
  float s = 0.f;
#pragma unroll
  for (int i = 0; i < 4; ++i) {
    s = fmaf(bf2f(pa[i] & 0xffffu), bf2f(pb[i] & 0xffffu), s);
    s = fmaf(bf2f(pa[i] >> 16),     bf2f(pb[i] >> 16),     s);
  }
  return s;
}

// ---------------------------------------------------------------------------
// f32 -> bf16 conversion, 8 elements per thread (vectorized b128 in/out)
// ---------------------------------------------------------------------------
__global__ __launch_bounds__(256)
void k_cvt_bf16(const float* __restrict__ src, unsigned short* __restrict__ dst, int n8) {
  int i = blockIdx.x * blockDim.x + threadIdx.x;
  if (i >= n8) return;
  const float4* s = (const float4*)src + (size_t)i * 2;
  float4 a = s[0], b = s[1];
  unsigned short o[8] = {f2bf(a.x), f2bf(a.y), f2bf(a.z), f2bf(a.w),
                         f2bf(b.x), f2bf(b.y), f2bf(b.z), f2bf(b.w)};
  *((uint4*)dst + i) = *(const uint4*)o;
}

// W [512, NC] f32 row-major -> Wt [NC, 512] bf16 (K-major). Once per weight.
__global__ __launch_bounds__(256)
void k_cvt_transpose_bf16(const float* __restrict__ W, unsigned short* __restrict__ Wt, int NC) {
  int idx = blockIdx.x * blockDim.x + threadIdx.x;
  if (idx >= NC * 512) return;
  int n = idx >> 9, k = idx & 511;
  Wt[idx] = f2bf(W[(size_t)k * NC + n]);
}

// cW2 [512,40] -> w2t [64,512] bf16 K-major (zero padded), cb2 -> f32[64]
__global__ void k_padw2t(const float* __restrict__ w2, const float* __restrict__ b2,
                         unsigned short* __restrict__ w2t, float* __restrict__ b2p) {
  int idx = blockIdx.x * blockDim.x + threadIdx.x;
  if (idx < 64 * 512) {
    int n = idx >> 9, k = idx & 511;
    w2t[idx] = f2bf((n < 40) ? w2[k * 40 + n] : 0.f);
  }
  if (idx < 64) b2p[idx] = (idx < 40) ? b2[idx] : 0.f;
}

// ---------------------------------------------------------------------------
// Tiled bf16 GEMM with WMMA:  C[M,NCOL] = A[M,512](bf16) x Bt[NCOL,512](bf16)^T
//   + bias (+ optional ReLU).  64x64 tile / WG, 8 waves = 2x4, 2 accs / wave.
// Double-buffered LDS; fills go through global_load_async_to_lds_b128
// (ASYNCcnt pipeline), with a sync b128 fallback.
// A is staged with contiguous K (no swap) so an A fragment (16x32 bf16) is two
// contiguous ds_load_b128 per lane; Bt is already K-major so a B fragment
// (32x16 bf16) needs no permutation at all.
// ---------------------------------------------------------------------------
template<int NCOL>
__global__ __launch_bounds__(256)
void k_gemm_wmma(const unsigned short* __restrict__ A,
                 const unsigned short* __restrict__ Bt,
                 const float* __restrict__ bias,
                 float* __restrict__ C, int M, int relu) {
  __shared__ __align__(64) unsigned short As[2][64 * 32];   // 2 x 4 KB
  __shared__ __align__(64) unsigned short Bs[2][64 * 32];   // 2 x 4 KB

  const int m0 = blockIdx.x * 64;
  const int n0 = blockIdx.y * 64;
  const int t    = threadIdx.x;
  const int lane = t & 31;
  const int w    = t >> 5;
  const int wm   = w >> 2;      // 0..1   (M sub-block of 32)
  const int wn   = w & 3;       // 0..3   (N sub-block of 16)

  v8f acc0 = {}, acc1 = {};

  // A fill: 64 rows x 32 K, 8 bf16 (16 B) per thread
  const int ar  = t >> 2;                  // 0..63 rows
  const int ac  = (t & 3) * 8;             // {0,8,16,24}
  const int apc = ac;                      // no K swap: contiguous K, matches B
  int arow = m0 + ar; if (arow >= M) arow = M - 1;       // clamp: EXEC stays full
  const unsigned short* agp = A + (size_t)arow * 512 + ac;

  // B fill: 64 cols x 32 K, 8 contiguous K (16 B) per thread from Bt[n][k]
  const int bn = t & 63;
  const int bk = (t >> 6) * 8;             // {0,8,16,24}
  const unsigned short* bgp = Bt + (size_t)(n0 + bn) * 512 + bk;

  const int fr = lane & 15;                // fragment row/col within 16
  const int fh = (lane >> 4) * 16;         // K half select (32 B chunk)

  auto fill = [&](int kk, int buf) {
#if USE_ASYNC_LDS
    __builtin_amdgcn_global_load_async_to_lds_b128(
        to_gas4(agp + kk * 32), to_las4(&As[buf][ar * 32 + apc]), 0, 0);
    __builtin_amdgcn_global_load_async_to_lds_b128(
        to_gas4(bgp + kk * 32), to_las4(&Bs[buf][bn * 32 + bk]), 0, 0);
#else
    *(uint4*)&As[buf][ar * 32 + apc] = *(const uint4*)(agp + kk * 32);
    *(uint4*)&Bs[buf][bn * 32 + bk]  = *(const uint4*)(bgp + kk * 32);
#endif
  };

  fill(0, 0);
  wait_async0();
  __syncthreads();

  for (int kk = 0; kk < 16; ++kk) {
    const int cur = kk & 1;
    if (kk < 15) fill(kk + 1, 1 - cur);    // async engine fills other buffer

    v16bf bf = *(const v16bf*)&Bs[cur][(wn * 16 + fr) * 32 + fh];
    v16bf a0 = *(const v16bf*)&As[cur][(wm * 32 +      fr) * 32 + fh];
    v16bf a1 = *(const v16bf*)&As[cur][(wm * 32 + 16 + fr) * 32 + fh];
    acc0 = __builtin_amdgcn_wmma_f32_16x16x32_bf16(false, a0, false, bf, (short)0, acc0, false, false);
    acc1 = __builtin_amdgcn_wmma_f32_16x16x32_bf16(false, a1, false, bf, (short)0, acc1, false, false);

    wait_async0();        // own async fills of next buffer complete
    __syncthreads();      // publish next buffer / retire reads of current
  }

  // Epilogue: C/D layout = VGPR i: lanes 0-15 -> M=i, lanes 16-31 -> M=i+8
  const int col = n0 + wn * 16 + fr;
  const float bv = bias[col];
  const int roff = (lane >> 4) * 8;
#pragma unroll
  for (int i = 0; i < 8; ++i) {
    int r0 = m0 + wm * 32 + roff + i;
    float v0 = acc0[i] + bv; if (relu) v0 = fmaxf(v0, 0.f);
    if (r0 < M) C[(size_t)r0 * NCOL + col] = v0;
    int r1 = r0 + 16;
    float v1 = acc1[i] + bv; if (relu) v1 = fmaxf(v1, 0.f);
    if (r1 < M) C[(size_t)r1 * NCOL + col] = v1;
  }
}

// ---------------------------------------------------------------------------
// One wave (32 lanes) per row LayerNorm over 512 columns (in place)
// ---------------------------------------------------------------------------
__global__ __launch_bounds__(256)
void k_layernorm(float* __restrict__ h, const float* __restrict__ g,
                 const float* __restrict__ b, int M) {
  int row = blockIdx.x * 8 + (threadIdx.x >> 5);
  int lane = threadIdx.x & 31;
  if (row >= M) return;
  float* hr = h + (size_t)row * 512;
  float s = 0.f, sq = 0.f;
#pragma unroll
  for (int j = 0; j < 16; ++j) { float v = hr[lane + 32 * j]; s += v; sq = fmaf(v, v, sq); }
  s = wred_sum(s); sq = wred_sum(sq);
  float mu  = s * (1.f / 512.f);
  float var = sq * (1.f / 512.f) - mu * mu;
  float inv = rsqrtf(var + 1e-5f);
#pragma unroll
  for (int j = 0; j < 16; ++j) {
    int c = lane + 32 * j;
    hr[c] = (hr[c] - mu) * inv * g[c] + b[c];
  }
}

// ---------------------------------------------------------------------------
// One wave per row: hn = bf16( h / (||h|| + 1e-4) ).  51 MB -> L2 resident.
// ---------------------------------------------------------------------------
__global__ __launch_bounds__(256)
void k_rownorm_bf16(const float* __restrict__ h, unsigned short* __restrict__ hn, int M) {
  int row = blockIdx.x * 8 + (threadIdx.x >> 5);
  int lane = threadIdx.x & 31;
  if (row >= M) return;
  const float4* hr = (const float4*)(h + (size_t)row * 512) + lane * 4;
  float4 v[4]; float ss = 0.f;
#pragma unroll
  for (int j = 0; j < 4; ++j) {
    v[j] = hr[j];
    ss += v[j].x * v[j].x + v[j].y * v[j].y + v[j].z * v[j].z + v[j].w * v[j].w;
  }
  ss = wred_sum(ss);
  float rn = 1.f / (sqrtf(ss) + 1e-4f);
  unsigned short o[16];
#pragma unroll
  for (int j = 0; j < 4; ++j) {
    o[j * 4 + 0] = f2bf(v[j].x * rn); o[j * 4 + 1] = f2bf(v[j].y * rn);
    o[j * 4 + 2] = f2bf(v[j].z * rn); o[j * 4 + 3] = f2bf(v[j].w * rn);
  }
  uint4* dst = (uint4*)(hn + (size_t)row * 512 + lane * 16);
  dst[0] = ((const uint4*)o)[0];
  dst[1] = ((const uint4*)o)[1];
}

// ---------------------------------------------------------------------------
// One wave per edge: cos = <hn[src],hn[dst]> (bf16 gathers out of L2),
// ang[src] += dinv[src]*dinv[dst]*clip(cos).
// ---------------------------------------------------------------------------
__global__ __launch_bounds__(256)
void k_edge_angle(const int* __restrict__ esrc, const int* __restrict__ edst,
                  const unsigned short* __restrict__ hn, const float* __restrict__ dinv,
                  float* __restrict__ ang, int E) {
  int e = blockIdx.x * 8 + (threadIdx.x >> 5);
  int lane = threadIdx.x & 31;
  if (e >= E) return;
  int s = esrc[e], d = edst[e];
  const uint4* rs = (const uint4*)(hn + (size_t)s * 512 + lane * 16);
  const uint4* rd = (const uint4*)(hn + (size_t)d * 512 + lane * 16);
  float acc = dot8_bf16(rs[0], rd[0]) + dot8_bf16(rs[1], rd[1]);
  acc = wred_sum(acc);
  if (lane == 0) {
    float c = fminf(1.f, fmaxf(-1.f, acc));
    atomicAdd(&ang[s], dinv[s] * dinv[d] * c);
  }
}

__global__ void k_rotate(float* __restrict__ h, const float* __restrict__ ang, int M) {
  int i = blockIdx.x * blockDim.x + threadIdx.x;
  if (i >= M) return;
  float a = ang[i];
  float c = cosf(a), s = sinf(a);
  float* hr = h + (size_t)i * 512;
  float h0 = hr[0], h1 = hr[1];
  hr[0] = c * h0 - s * h1;
  hr[1] = s * h0 + c * h1;
}

__global__ void k_zero_f32(float* __restrict__ p, int n) {
  int i = blockIdx.x * blockDim.x + threadIdx.x;
  if (i < n) p[i] = 0.f;
}
__global__ void k_deg(const int* __restrict__ esrc, float* __restrict__ deg, int E) {
  int i = blockIdx.x * blockDim.x + threadIdx.x;
  if (i < E) atomicAdd(&deg[esrc[i]], 1.f);
}
__global__ void k_dinv(float* __restrict__ deg, int M) {
  int i = blockIdx.x * blockDim.x + threadIdx.x;
  if (i < M) { float d = deg[i]; deg[i] = (d > 0.f) ? rsqrtf(d) : 0.f; }
}

__global__ __launch_bounds__(256)
void k_bn(float* __restrict__ z, const float* __restrict__ g, const float* __restrict__ b,
          const float* __restrict__ mean, const float* __restrict__ var, int n) {
  int i = blockIdx.x * blockDim.x + threadIdx.x;
  if (i >= n) return;
  int c = i & 511;
  z[i] = (z[i] - mean[c]) * rsqrtf(var[c] + 1e-5f) * g[c] + b[c];
}

// One wave per row: log_softmax over 40 logits stored in a 64-wide buffer
__global__ __launch_bounds__(256)
void k_logsoftmax(const float* __restrict__ lg, float* __restrict__ out, int M) {
  int row = blockIdx.x * 8 + (threadIdx.x >> 5);
  int lane = threadIdx.x & 31;
  if (row >= M) return;
  const float* r = lg + (size_t)row * 64;
  float v0 = r[lane];
  int c1 = lane + 32;
  float v1 = (c1 < 40) ? r[c1] : -3.4e38f;
  float m = wred_max(fmaxf(v0, v1));
  float e = expf(v0 - m) + ((c1 < 40) ? expf(v1 - m) : 0.f);
  float ssum = wred_sum(e);
  float ls = logf(ssum);
  out[(size_t)row * 40 + lane] = v0 - m - ls;
  if (c1 < 40) out[(size_t)row * 40 + c1] = v1 - m - ls;
}

// ---------------------------------------------------------------------------
extern "C" void kernel_launch(void* const* d_in, const int* in_sizes, int n_in,
                              void* d_out, int out_size, void* d_ws, size_t ws_size,
                              hipStream_t stream) {
  const float* x     = (const float*)d_in[0];
  const int*   esrc  = (const int*)  d_in[1];
  const int*   edst  = (const int*)  d_in[2];
  const float* W_in  = (const float*)d_in[3];
  const float* b_in  = (const float*)d_in[4];
  const float* ln_g  = (const float*)d_in[5];
  const float* ln_b  = (const float*)d_in[6];
  const float* cW1   = (const float*)d_in[7];
  const float* cb1   = (const float*)d_in[8];
  const float* bn_g  = (const float*)d_in[9];
  const float* bn_b  = (const float*)d_in[10];
  const float* bn_m  = (const float*)d_in[11];
  const float* bn_v  = (const float*)d_in[12];
  const float* cW2   = (const float*)d_in[13];
  const float* cb2   = (const float*)d_in[14];
  float* out = (float*)d_out;

  const int N = in_sizes[0] / 512;   // 50000
  const int E = in_sizes[1];         // 1600000
  const int ND = N * 512;

  // Workspace carve-up (≈257 MB)
  char* base = (char*)d_ws;
  size_t cur = 0;
  auto take = [&](size_t bytes) -> void* {
    void* p = base + cur;
    cur += (bytes + 255) & ~(size_t)255;
    return p;
  };
  float*          h    = (float*)         take((size_t)ND * 4);
  unsigned short* hbf  = (unsigned short*)take((size_t)ND * 2);
  float*          z    = (float*)         take((size_t)ND * 4);   // reused for padded logits
  unsigned short* wbf  = (unsigned short*)take((size_t)512 * 512 * 2);  // K-major weights
  unsigned short* w2bf = (unsigned short*)take((size_t)64 * 512 * 2);   // K-major padded cW2
  float*          b2p  = (float*)         take(64 * 4);
  float*          dinv = (float*)         take((size_t)N * 4);
  float*          ang  = (float*)         take((size_t)N * 4);
  if (cur > ws_size) return;

  const int mtiles = (N + 63) / 64;                  // 782
  const int rowblk = (N + 7) / 8;                    // wave-per-row kernels
  auto g1 = [](int n, int b) { return (n + b - 1) / b; };

  // 1) x -> bf16, W_in -> bf16 K-major; GEMM1 (+bias+ReLU); LayerNorm
  k_cvt_bf16<<<g1(ND / 8, 256), 256, 0, stream>>>(x, hbf, ND / 8);
  k_cvt_transpose_bf16<<<g1(512 * 512, 256), 256, 0, stream>>>(W_in, wbf, 512);
  k_gemm_wmma<512><<<dim3(mtiles, 8), 256, 0, stream>>>(hbf, wbf, b_in, h, N, 1);
  k_layernorm<<<rowblk, 256, 0, stream>>>(h, ln_g, ln_b, N);

  // 2) degrees -> dinv
  k_zero_f32<<<g1(N, 256), 256, 0, stream>>>(dinv, N);
  k_deg<<<g1(E, 256), 256, 0, stream>>>(esrc, dinv, E);
  k_dinv<<<g1(N, 256), 256, 0, stream>>>(dinv, N);

  // 3) three angle-rotation layers (hn bf16 stays L2-resident: 51 MB < 192 MB)
  for (int l = 0; l < 3; ++l) {
    k_rownorm_bf16<<<rowblk, 256, 0, stream>>>(h, hbf, N);
    k_zero_f32<<<g1(N, 256), 256, 0, stream>>>(ang, N);
    k_edge_angle<<<g1(E, 8), 256, 0, stream>>>(esrc, edst, hbf, dinv, ang, E);
    k_rotate<<<g1(N, 256), 256, 0, stream>>>(h, ang, N);
  }

  // 4) classifier GEMM1 (+bias+ReLU) then BatchNorm (eval)
  k_cvt_transpose_bf16<<<g1(512 * 512, 256), 256, 0, stream>>>(cW1, wbf, 512);
  k_cvt_bf16<<<g1(ND / 8, 256), 256, 0, stream>>>(h, hbf, ND / 8);
  k_gemm_wmma<512><<<dim3(mtiles, 8), 256, 0, stream>>>(hbf, wbf, cb1, z, N, 1);
  k_bn<<<g1(ND, 256), 256, 0, stream>>>(z, bn_g, bn_b, bn_m, bn_v, ND);

  // 5) classifier GEMM2 via padded K-major [64,512] bf16 weights, log_softmax
  k_cvt_bf16<<<g1(ND / 8, 256), 256, 0, stream>>>(z, hbf, ND / 8);
  k_padw2t<<<g1(64 * 512, 256), 256, 0, stream>>>(cW2, cb2, w2bf, b2p);
  k_gemm_wmma<64><<<dim3(mtiles, 1), 256, 0, stream>>>(hbf, w2bf, b2p, z, N, 0);
  k_logsoftmax<<<rowblk, 256, 0, stream>>>(z, out, N);
}
